// DefaultSegmentLinear_80762565034002
// MI455X (gfx1250) — 8-bit matrix operand order repaired and checked by reading, compiled, not run on hardware
//
#include <hip/hip_runtime.h>
#include <hip/hip_bf16.h>


typedef int v8i __attribute__((ext_vector_type(8)));

#define M_DIM 16384
#define N_DIM 4096
#define K_DIM 4096
#define CHUNK_D 1024
#define NCHUNK 4
#define QMAXF 127.0f
#define EPSF 1e-8f

#define QX_BYTES ((size_t)M_DIM * K_DIM)
#define QW_BYTES ((size_t)N_DIM * K_DIM)
#define STATS_WORDS 16

__global__ void k_init_stats(unsigned* stats) {
    if (threadIdx.x < STATS_WORDS) stats[threadIdx.x] = 0u;
}

__global__ void k_absmax_w(const float* __restrict__ w, unsigned* stats, int n4) {
    __shared__ unsigned smax;
    if (threadIdx.x == 0) smax = 0u;
    __syncthreads();
    unsigned lm = 0u;
    int stride = gridDim.x * blockDim.x;
    for (int i = blockIdx.x * blockDim.x + threadIdx.x; i < n4; i += stride) {
        float4 v = ((const float4*)w)[i];
        unsigned b0 = __float_as_uint(__builtin_fabsf(v.x));
        unsigned b1 = __float_as_uint(__builtin_fabsf(v.y));
        unsigned b2 = __float_as_uint(__builtin_fabsf(v.z));
        unsigned b3 = __float_as_uint(__builtin_fabsf(v.w));
        lm = max(lm, max(max(b0, b1), max(b2, b3)));
    }
    atomicMax(&smax, lm);
    __syncthreads();
    if (threadIdx.x == 0) atomicMax(&stats[0], smax);
}

__global__ void k_absmax_x(const float* __restrict__ x, unsigned* stats, int n4) {
    __shared__ unsigned smax[NCHUNK];
    if (threadIdx.x < NCHUNK) smax[threadIdx.x] = 0u;
    __syncthreads();
    unsigned lm[NCHUNK] = {0u, 0u, 0u, 0u};
    int stride = gridDim.x * blockDim.x;
    for (int i = blockIdx.x * blockDim.x + threadIdx.x; i < n4; i += stride) {
        int k = (i << 2) & (K_DIM - 1);
        int c = k >> 10;
        float4 v = ((const float4*)x)[i];
        unsigned b0 = __float_as_uint(__builtin_fabsf(v.x));
        unsigned b1 = __float_as_uint(__builtin_fabsf(v.y));
        unsigned b2 = __float_as_uint(__builtin_fabsf(v.z));
        unsigned b3 = __float_as_uint(__builtin_fabsf(v.w));
        lm[c] = max(lm[c], max(max(b0, b1), max(b2, b3)));
    }
#pragma unroll
    for (int c = 0; c < NCHUNK; ++c)
        if (lm[c]) atomicMax(&smax[c], lm[c]);
    __syncthreads();
    if (threadIdx.x < NCHUNK) atomicMax(&stats[1 + threadIdx.x], smax[threadIdx.x]);
}

__global__ void k_finalize(unsigned* stats) {
    if (threadIdx.x == 0 && blockIdx.x == 0) {
        float* f = (float*)stats;
        float sw = fmaxf(__uint_as_float(stats[0]) / QMAXF, EPSF);
        f[5] = 1.0f / sw;
#pragma unroll
        for (int c = 0; c < NCHUNK; ++c) {
            float sx = fmaxf(__uint_as_float(stats[1 + c]) / QMAXF, EPSF);
            f[6 + c] = 1.0f / sx;
            f[10 + c] = sw * sx;
        }
    }
}

__device__ __forceinline__ int quant_pack4(float4 v, float inv) {
    int q0 = min(127, max(-127, __float2int_rn(v.x * inv)));
    int q1 = min(127, max(-127, __float2int_rn(v.y * inv)));
    int q2 = min(127, max(-127, __float2int_rn(v.z * inv)));
    int q3 = min(127, max(-127, __float2int_rn(v.w * inv)));
    return (q0 & 0xFF) | ((q1 & 0xFF) << 8) | ((q2 & 0xFF) << 16) | ((q3 & 0xFF) << 24);
}

__global__ void k_quant_w(const float* __restrict__ w, int* __restrict__ qw,
                          const unsigned* __restrict__ stats, int n4) {
    float inv = ((const float*)stats)[5];
    int stride = gridDim.x * blockDim.x;
    for (int i = blockIdx.x * blockDim.x + threadIdx.x; i < n4; i += stride)
        qw[i] = quant_pack4(((const float4*)w)[i], inv);
}

__global__ void k_quant_x(const float* __restrict__ x, int* __restrict__ qx,
                          const unsigned* __restrict__ stats, int n4) {
    const float* f = (const float*)stats;
    float inv[NCHUNK] = {f[6], f[7], f[8], f[9]};
    int stride = gridDim.x * blockDim.x;
    for (int i = blockIdx.x * blockDim.x + threadIdx.x; i < n4; i += stride) {
        int k = (i << 2) & (K_DIM - 1);
        int c = k >> 10;
        qx[i] = quant_pack4(((const float4*)x)[i], inv[c]);
    }
}

__device__ __forceinline__ v8i load_a_frag(const signed char* p) {
    int2 a0 = *(const int2*)(p);
    int2 a1 = *(const int2*)(p + 16);
    int2 a2 = *(const int2*)(p + 32);
    int2 a3 = *(const int2*)(p + 48);
    v8i a;
    a[0] = a0.x; a[1] = a0.y; a[2] = a1.x; a[3] = a1.y;
    a[4] = a2.x; a[5] = a2.y; a[6] = a3.x; a[7] = a3.y;
    return a;
}
__device__ __forceinline__ v8i load_b_frag(const signed char* p) {
    int2 b0 = *(const int2*)(p);
    int2 b1 = *(const int2*)(p + 16);
    int2 b2 = *(const int2*)(p + 32);
    int2 b3 = *(const int2*)(p + 48);
    v8i b;
    b[0] = b0.x; b[1] = b0.y; b[2] = b1.x; b[3] = b1.y;
    b[4] = b2.x; b[5] = b2.y; b[6] = b3.x; b[7] = b3.y;
    return b;
}

__global__ __launch_bounds__(256) void k_gemm_i8(
    const signed char* __restrict__ qx, const signed char* __restrict__ qw,
    const unsigned* __restrict__ statsu, const float* __restrict__ bias,
    float* __restrict__ out) {
    const float* stats = (const float*)statsu;
    const int tid = threadIdx.x;
    const int wid = tid >> 5;
    const int lane = tid & 31;
    const int wm = wid >> 1;
    const int wn = wid & 1;
    const int m0 = blockIdx.y * 128 + wm * 32;
    const int n0 = blockIdx.x * 128 + wn * 64;
    const int r = lane & 15;
    const int kh = lane >> 4;

    const signed char* pa0 = qx + (size_t)(m0 + r) * K_DIM + kh * 8;
    const signed char* pa1 = pa0 + (size_t)16 * K_DIM;
    const signed char* pb = qw + (size_t)(n0 + r) * K_DIM + kh * 8;

    const float csc0 = stats[10], csc1 = stats[11], csc2 = stats[12], csc3 = stats[13];
    float facc[2][4][8];
#pragma unroll
    for (int mt = 0; mt < 2; ++mt)
#pragma unroll
        for (int nt = 0; nt < 4; ++nt)
#pragma unroll
            for (int e = 0; e < 8; ++e) facc[mt][nt][e] = 0.0f;

#pragma unroll 1
    for (int c = 0; c < NCHUNK; ++c) {
        v8i iacc[2][4];
#pragma unroll
        for (int mt = 0; mt < 2; ++mt)
#pragma unroll
            for (int nt = 0; nt < 4; ++nt) iacc[mt][nt] = (v8i){};
        const int kbase = c * CHUNK_D;
#pragma unroll 2
        for (int kk = 0; kk < CHUNK_D; kk += 64) {
            const int k = kbase + kk;
            __builtin_prefetch(pa0 + k + 64, 0, 3);
            __builtin_prefetch(pa1 + k + 64, 0, 3);
            __builtin_prefetch(pb + k + 64, 0, 3);
            v8i a0 = load_a_frag(pa0 + k);
            v8i a1 = load_a_frag(pa1 + k);
            v8i b0 = load_b_frag(pb + k);
            v8i b1 = load_b_frag(pb + k + (size_t)16 * K_DIM);
            v8i b2 = load_b_frag(pb + k + (size_t)32 * K_DIM);
            v8i b3 = load_b_frag(pb + k + (size_t)48 * K_DIM);
            iacc[0][0] = __builtin_amdgcn_wmma_i32_16x16x64_iu8(true, a0, true, b0, iacc[0][0], false, false);
            iacc[0][1] = __builtin_amdgcn_wmma_i32_16x16x64_iu8(true, a0, true, b1, iacc[0][1], false, false);
            iacc[0][2] = __builtin_amdgcn_wmma_i32_16x16x64_iu8(true, a0, true, b2, iacc[0][2], false, false);
            iacc[0][3] = __builtin_amdgcn_wmma_i32_16x16x64_iu8(true, a0, true, b3, iacc[0][3], false, false);
            iacc[1][0] = __builtin_amdgcn_wmma_i32_16x16x64_iu8(true, a1, true, b0, iacc[1][0], false, false);
            iacc[1][1] = __builtin_amdgcn_wmma_i32_16x16x64_iu8(true, a1, true, b1, iacc[1][1], false, false);
            iacc[1][2] = __builtin_amdgcn_wmma_i32_16x16x64_iu8(true, a1, true, b2, iacc[1][2], false, false);
            iacc[1][3] = __builtin_amdgcn_wmma_i32_16x16x64_iu8(true, a1, true, b3, iacc[1][3], false, false);
        }
        const float s = (c == 0) ? csc0 : (c == 1) ? csc1 : (c == 2) ? csc2 : csc3;
#pragma unroll
        for (int mt = 0; mt < 2; ++mt)
#pragma unroll
            for (int nt = 0; nt < 4; ++nt)
#pragma unroll
                for (int e = 0; e < 8; ++e)
                    facc[mt][nt][e] += (float)iacc[mt][nt][e] * s;
    }

    const int col = lane & 15;
    const int rofs = 8 * kh;
    float bv[4];
#pragma unroll
    for (int nt = 0; nt < 4; ++nt) bv[nt] = bias[n0 + nt * 16 + col];
#pragma unroll
    for (int mt = 0; mt < 2; ++mt) {
#pragma unroll
        for (int e = 0; e < 8; ++e) {
            const size_t row = (size_t)(m0 + mt * 16 + rofs + e);
#pragma unroll
            for (int nt = 0; nt < 4; ++nt)
                out[row * N_DIM + n0 + nt * 16 + col] = facc[mt][nt][e] + bv[nt];
        }
    }
}

extern "C" void kernel_launch(void* const* d_in, const int* in_sizes, int n_in,
                              void* d_out, int out_size, void* d_ws, size_t ws_size,
                              hipStream_t stream) {
    const float* x = (const float*)d_in[0];
    const float* w = (const float*)d_in[1];
    const float* bias = (const float*)d_in[2];
    float* out = (float*)d_out;

    char* ws = (char*)d_ws;
    signed char* qx = (signed char*)ws;
    signed char* qw = (signed char*)(ws + QX_BYTES);
    unsigned* stats = (unsigned*)(ws + QX_BYTES + QW_BYTES);

    const int wn4 = (N_DIM * K_DIM) / 4;
    const int xn4 = (M_DIM * K_DIM) / 4;

    k_init_stats<<<1, 32, 0, stream>>>(stats);
    k_absmax_w<<<1024, 256, 0, stream>>>(w, stats, wn4);
    k_absmax_x<<<4096, 256, 0, stream>>>(x, stats, xn4);
    k_finalize<<<1, 32, 0, stream>>>(stats);
    k_quant_w<<<2048, 256, 0, stream>>>(w, (int*)qw, stats, wn4);
    k_quant_x<<<8192, 256, 0, stream>>>(x, (int*)qx, stats, xn4);

    dim3 grid(N_DIM / 128, M_DIM / 128);
    k_gemm_i8<<<grid, 256, 0, stream>>>(qx, qw, stats, bias, out);
}
